// Attention_29600914604226
// MI455X (gfx1250) — matrix operand order repaired and checked by reading, not compiled, not run on hardware
//
#include <hip/hip_runtime.h>
#include <hip/hip_bf16.h>

typedef __bf16 bf16_t;
typedef __attribute__((ext_vector_type(16))) __bf16 v16bf;
typedef __attribute__((ext_vector_type(8)))  __bf16 v8bf;
typedef __attribute__((ext_vector_type(4)))  __bf16 v4bf;
typedef __attribute__((ext_vector_type(8)))  float  v8f;
typedef __attribute__((ext_vector_type(4)))  float  v4f;

#define S_LEN   8192
#define D_MODEL 1024
#define D_HEAD  128
#define NEG_INF (-1e9f)

// ---------------------------------------------------------------------------
// CDNA5 wave32 WMMA fragment layouts (ISA 7.12.2):
//  A (16x32 bf16): lane<16  -> row=lane,    K = kb+{0..7} and kb+{16..23}
//                  lane>=16 -> row=lane-16, K = kb+{8..15} and kb+{24..31}
//  B (32x16 bf16): same K split as A, for column n=lane&15:
//                  lanes 0-15: K = kb+{0..7},{16..23} ; lanes 16-31: K = kb+{8..15},{24..31}
//  C/D (16x16 f32): acc[r] at lane L -> element (M = r + 8*(L>=16), N = L&15)
// ---------------------------------------------------------------------------
__device__ __forceinline__ v8f wmma_bf16(v16bf a, v16bf b, v8f c) {
  return __builtin_amdgcn_wmma_f32_16x16x32_bf16(false, a, false, b, (short)0, c,
                                                 false, false);
}

// A fragment from row-major [*, ld] source.
__device__ __forceinline__ v16bf load_a(const bf16_t* __restrict__ base, int ld,
                                        int row, int kb, int hf) {
  union { v16bf v; v8bf h[2]; } u;
  const bf16_t* p = base + (size_t)row * ld + kb + 8 * hf;
  u.h[0] = *(const v8bf*)(p);       // K = kb + 8hf .. +7
  u.h[1] = *(const v8bf*)(p + 16);  // K = kb + 16 + 8hf .. +7
  return u.v;
}

// B fragment: source is B^T stored row-major (row = output column, ld over K).
__device__ __forceinline__ v16bf load_b(const bf16_t* __restrict__ base, int ld,
                                        int col, int kb, int hf) {
  union { v16bf v; v8bf h[2]; } u;
  const bf16_t* p = base + (size_t)col * ld + kb + 8 * hf;
  u.h[0] = *(const v8bf*)(p);       // K = kb + 8hf .. +7
  u.h[1] = *(const v8bf*)(p + 16);  // K = kb + 16 + 8hf .. +7
  return u.v;
}

// ---------------------------------------------------------------------------
// fp32 -> bf16 elementwise convert (vectorized x4, streaming NT loads)
// ---------------------------------------------------------------------------
__global__ __launch_bounds__(256) void cvt_kernel(const float* __restrict__ in,
                                                  bf16_t* __restrict__ out, int n4) {
  int i = blockIdx.x * blockDim.x + threadIdx.x;
  if (i < n4) {
    v4f f = __builtin_nontemporal_load(((const v4f*)in) + i);
    v4bf o;
    o.x = (bf16_t)f.x; o.y = (bf16_t)f.y; o.z = (bf16_t)f.z; o.w = (bf16_t)f.w;
    ((v4bf*)out)[i] = o;
  }
}

// fp32 [rows][cols] -> bf16 transposed [cols][rows]
__global__ __launch_bounds__(256) void cvt_t_kernel(const float* __restrict__ in,
                                                    bf16_t* __restrict__ out,
                                                    int rows, int cols) {
  int idx = blockIdx.x * blockDim.x + threadIdx.x;
  if (idx < rows * cols) {
    int r = idx / cols, c = idx % cols;
    out[c * rows + r] = (bf16_t)in[idx];
  }
}

// ---------------------------------------------------------------------------
// Projection GEMM: [S,1024](bf16) x [1024,128] -> [S,128] bf16.
// WT pre-transposed to [128][1024]. One wave per 32x32 tile (2x2 WMMA tiles,
// 8 loads : 4 wmma per k-step). TRANS stores transposed [128][S] (for V).
// TRANS is a template arg so each instantiation has a straight-line epilogue.
// ---------------------------------------------------------------------------
template <int TRANS>
__global__ __launch_bounds__(256) void proj_kernel(const bf16_t* __restrict__ A,
                                                   const bf16_t* __restrict__ WT,
                                                   bf16_t* __restrict__ out) {
  int wave = (blockIdx.x * blockDim.x + threadIdx.x) >> 5;
  int lane = threadIdx.x & 31;
  int hf   = lane >> 4;
  int l16  = lane & 15;

  const int ntn = D_HEAD / 32;  // 4 column strips
  int r0 = (wave / ntn) * 32;
  int c0 = (wave % ntn) * 32;

  v8f acc[2][2];
#pragma unroll
  for (int i = 0; i < 2; ++i)
#pragma unroll
    for (int j = 0; j < 2; ++j)
      acc[i][j] = (v8f){0.f, 0.f, 0.f, 0.f, 0.f, 0.f, 0.f, 0.f};

#pragma unroll 2
  for (int kb = 0; kb < D_MODEL; kb += 32) {
    v16bf a0 = load_a(A, D_MODEL, r0 + l16, kb, hf);
    v16bf a1 = load_a(A, D_MODEL, r0 + 16 + l16, kb, hf);
    v16bf b0 = load_b(WT, D_MODEL, c0 + l16, kb, hf);
    v16bf b1 = load_b(WT, D_MODEL, c0 + 16 + l16, kb, hf);
    acc[0][0] = wmma_bf16(a0, b0, acc[0][0]);
    acc[0][1] = wmma_bf16(a0, b1, acc[0][1]);
    acc[1][0] = wmma_bf16(a1, b0, acc[1][0]);
    acc[1][1] = wmma_bf16(a1, b1, acc[1][1]);
  }

#pragma unroll
  for (int i = 0; i < 2; ++i)
#pragma unroll
    for (int j = 0; j < 2; ++j)
#pragma unroll
      for (int r = 0; r < 8; ++r) {
        int row = r0 + 16 * i + r + 8 * hf;
        int col = c0 + 16 * j + l16;
        if (TRANS) out[(size_t)col * S_LEN + row]  = (bf16_t)acc[i][j][r];
        else       out[(size_t)row * D_HEAD + col] = (bf16_t)acc[i][j][r];
      }
}

// ---------------------------------------------------------------------------
// Flash attention (causal). One block (4 waves) per 16-row query tile.
// Waves split the key-block range round-robin (online softmax is order-free),
// then merge partial (m, l, O) through LDS.
// ---------------------------------------------------------------------------
#define NWARPS 4
__global__ __launch_bounds__(128) void flash_kernel(const bf16_t* __restrict__ qp,
                                                    const bf16_t* __restrict__ kp,
                                                    const bf16_t* __restrict__ vpT,
                                                    bf16_t* __restrict__ obf) {
  __shared__ bf16_t pbuf[NWARPS][16 * 32];        // P staging (C-layout -> A-layout)
  __shared__ float  oShare[NWARPS][16][D_HEAD];   // per-wave partial O
  __shared__ float  mShare[NWARPS][16];
  __shared__ float  lShare[NWARPS][16];

  int w    = threadIdx.x >> 5;
  int lane = threadIdx.x & 31;
  int hf   = lane >> 4;
  int l16  = lane & 15;

  int qb = blockIdx.x * 16;                       // first query row of this tile

  // Q fragments for the full K=128 head dim (loaded once, 32 VGPRs)
  v16bf qa[4];
#pragma unroll
  for (int kk = 0; kk < 4; ++kk)
    qa[kk] = load_a(qp, D_HEAD, qb + l16, kk * 32, hf);

  v8f o[8];
#pragma unroll
  for (int t = 0; t < 8; ++t) o[t] = (v8f){0.f, 0.f, 0.f, 0.f, 0.f, 0.f, 0.f, 0.f};
  float mrun[8], lrun[8];
#pragma unroll
  for (int r = 0; r < 8; ++r) { mrun[r] = -1e30f; lrun[r] = 0.f; }

  const float scale = 0.08838834764831845f;       // 1/sqrt(128)
  int kmax = qb + 16;                             // keys [0, qb+16) may be visible

  for (int kb = w * 32; kb < kmax; kb += NWARPS * 32) {
    // prefetch this wave's next key block
    __builtin_prefetch(kp + (size_t)(kb + NWARPS * 32) * D_HEAD, 0, 1);

    // ---- S = Qp * Kp^T : two 16x16 tiles (keys kb..kb+15, kb+16..kb+31)
    v8f s0 = {0.f, 0.f, 0.f, 0.f, 0.f, 0.f, 0.f, 0.f};
    v8f s1 = {0.f, 0.f, 0.f, 0.f, 0.f, 0.f, 0.f, 0.f};
#pragma unroll
    for (int kk = 0; kk < 4; ++kk) {
      v16bf b0 = load_b(kp, D_HEAD, kb + l16, kk * 32, hf);
      v16bf b1 = load_b(kp, D_HEAD, kb + 16 + l16, kk * 32, hf);
      s0 = wmma_bf16(qa[kk], b0, s0);
      s1 = wmma_bf16(qa[kk], b1, s1);
    }

    bool needMask = (kb + 31 > qb);
    float pr0[8], pr1[8];
#pragma unroll
    for (int r = 0; r < 8; ++r) {
      float v0 = s0[r] * scale;
      float v1 = s1[r] * scale;
      if (needMask) {
        int i = qb + r + 8 * hf;
        if (kb + l16 > i)      v0 = NEG_INF;
        if (kb + 16 + l16 > i) v1 = NEG_INF;
      }
      // row reduction across the 16 lanes of this half (rows r + 8*hf)
      float mx = fmaxf(v0, v1);
      mx = fmaxf(mx, __shfl_xor(mx, 1, 32));
      mx = fmaxf(mx, __shfl_xor(mx, 2, 32));
      mx = fmaxf(mx, __shfl_xor(mx, 4, 32));
      mx = fmaxf(mx, __shfl_xor(mx, 8, 32));
      float mnew  = fmaxf(mrun[r], mx);
      float alpha = __expf(mrun[r] - mnew);
      float e0 = __expf(v0 - mnew);
      float e1 = __expf(v1 - mnew);
      float rs = e0 + e1;
      rs += __shfl_xor(rs, 1, 32);
      rs += __shfl_xor(rs, 2, 32);
      rs += __shfl_xor(rs, 4, 32);
      rs += __shfl_xor(rs, 8, 32);
      lrun[r] = lrun[r] * alpha + rs;
      mrun[r] = mnew;
      pr0[r] = e0;
      pr1[r] = e1;
#pragma unroll
      for (int t = 0; t < 8; ++t) o[t][r] *= alpha;
    }

    // ---- P (C-layout) -> LDS -> A fragment (wave-private; DS ops in order)
#pragma unroll
    for (int r = 0; r < 8; ++r) {
      pbuf[w][(r + 8 * hf) * 32 + l16]      = (bf16_t)pr0[r];
      pbuf[w][(r + 8 * hf) * 32 + 16 + l16] = (bf16_t)pr1[r];
    }
    v16bf pa;
    {
      union { v16bf v; v8bf h[2]; } u;
      const bf16_t* p = &pbuf[w][l16 * 32 + 8 * hf];
      u.h[0] = *(const v8bf*)(p);
      u.h[1] = *(const v8bf*)(p + 16);
      pa = u.v;
    }

    // ---- O += P * Vp   (Vp stored transposed [128][S] for contiguous B frags)
#pragma unroll
    for (int t = 0; t < 8; ++t) {
      v16bf bv = load_b(vpT, S_LEN, t * 16 + l16, kb, hf);
      o[t] = wmma_bf16(pa, bv, o[t]);
    }
  }

  // ---- publish partials and merge across the 4 waves
#pragma unroll
  for (int t = 0; t < 8; ++t)
#pragma unroll
    for (int r = 0; r < 8; ++r)
      oShare[w][r + 8 * hf][t * 16 + l16] = o[t][r];
  if (l16 == 0) {
#pragma unroll
    for (int r = 0; r < 8; ++r) {
      mShare[w][r + 8 * hf] = mrun[r];
      lShare[w][r + 8 * hf] = lrun[r];
    }
  }
  __syncthreads();

  int c = threadIdx.x;  // 128 threads -> 128 head-dim columns
  for (int i = 0; i < 16; ++i) {
    float M = mShare[0][i];
#pragma unroll
    for (int ww = 1; ww < NWARPS; ++ww) M = fmaxf(M, mShare[ww][i]);
    float L = 0.f, acc = 0.f;
#pragma unroll
    for (int ww = 0; ww < NWARPS; ++ww) {
      float sc = __expf(mShare[ww][i] - M);
      L   += lShare[ww][i] * sc;
      acc += oShare[ww][i][c] * sc;
    }
    obf[(size_t)(qb + i) * D_HEAD + c] = (bf16_t)(acc / L);
  }
}

// ---------------------------------------------------------------------------
// Output GEMM: [S,128](bf16) x [128,1024] -> fp32 d_out (streamed NT stores).
// woT pre-transposed to [1024][128]. One wave per 32x32 tile.
// ---------------------------------------------------------------------------
__global__ __launch_bounds__(256) void out_gemm(const bf16_t* __restrict__ obf,
                                                const bf16_t* __restrict__ woT,
                                                float* __restrict__ out) {
  int wave = (blockIdx.x * blockDim.x + threadIdx.x) >> 5;
  int lane = threadIdx.x & 31;
  int hf   = lane >> 4;
  int l16  = lane & 15;

  const int ntn = D_MODEL / 32;  // 32 column strips
  int r0 = (wave / ntn) * 32;
  int c0 = (wave % ntn) * 32;

  v8f acc[2][2];
#pragma unroll
  for (int i = 0; i < 2; ++i)
#pragma unroll
    for (int j = 0; j < 2; ++j)
      acc[i][j] = (v8f){0.f, 0.f, 0.f, 0.f, 0.f, 0.f, 0.f, 0.f};

#pragma unroll
  for (int kb = 0; kb < D_HEAD; kb += 32) {
    v16bf a0 = load_a(obf, D_HEAD, r0 + l16, kb, hf);
    v16bf a1 = load_a(obf, D_HEAD, r0 + 16 + l16, kb, hf);
    v16bf b0 = load_b(woT, D_HEAD, c0 + l16, kb, hf);
    v16bf b1 = load_b(woT, D_HEAD, c0 + 16 + l16, kb, hf);
    acc[0][0] = wmma_bf16(a0, b0, acc[0][0]);
    acc[0][1] = wmma_bf16(a0, b1, acc[0][1]);
    acc[1][0] = wmma_bf16(a1, b0, acc[1][0]);
    acc[1][1] = wmma_bf16(a1, b1, acc[1][1]);
  }

#pragma unroll
  for (int i = 0; i < 2; ++i)
#pragma unroll
    for (int j = 0; j < 2; ++j)
#pragma unroll
      for (int r = 0; r < 8; ++r)
        __builtin_nontemporal_store(
            acc[i][j][r],
            out + (size_t)(r0 + 16 * i + r + 8 * hf) * D_MODEL + c0 + 16 * j + l16);
}

// ---------------------------------------------------------------------------
extern "C" void kernel_launch(void* const* d_in, const int* in_sizes, int n_in,
                              void* d_out, int out_size, void* d_ws, size_t ws_size,
                              hipStream_t stream) {
  const float* q   = (const float*)d_in[0];
  const float* k   = (const float*)d_in[1];
  const float* v   = (const float*)d_in[2];
  const float* w_q = (const float*)d_in[3];
  const float* w_k = (const float*)d_in[4];
  const float* w_v = (const float*)d_in[5];
  const float* w_o = (const float*)d_in[6];
  float* out = (float*)d_out;

  char* ws = (char*)d_ws;
  size_t off = 0;
  auto alloc = [&](size_t bytes) {
    char* p = ws + off;
    off += (bytes + 255) & ~(size_t)255;
    return p;
  };

  bf16_t* qbf = (bf16_t*)alloc((size_t)S_LEN * D_MODEL * 2);
  bf16_t* kbf = (bf16_t*)alloc((size_t)S_LEN * D_MODEL * 2);
  bf16_t* vbf = (bf16_t*)alloc((size_t)S_LEN * D_MODEL * 2);
  bf16_t* wqT = (bf16_t*)alloc((size_t)D_HEAD * D_MODEL * 2);
  bf16_t* wkT = (bf16_t*)alloc((size_t)D_HEAD * D_MODEL * 2);
  bf16_t* wvT = (bf16_t*)alloc((size_t)D_HEAD * D_MODEL * 2);
  bf16_t* woT = (bf16_t*)alloc((size_t)D_MODEL * D_HEAD * 2);
  bf16_t* qp  = (bf16_t*)alloc((size_t)S_LEN * D_HEAD * 2);
  bf16_t* kp  = (bf16_t*)alloc((size_t)S_LEN * D_HEAD * 2);
  bf16_t* vpT = (bf16_t*)alloc((size_t)D_HEAD * S_LEN * 2);
  bf16_t* obf = (bf16_t*)alloc((size_t)S_LEN * D_HEAD * 2);

  // 1) convert activations to bf16
  {
    int n4 = (S_LEN * D_MODEL) / 4;  // 2M
    int grid = (n4 + 255) / 256;
    cvt_kernel<<<grid, 256, 0, stream>>>(q, qbf, n4);
    cvt_kernel<<<grid, 256, 0, stream>>>(k, kbf, n4);
    cvt_kernel<<<grid, 256, 0, stream>>>(v, vbf, n4);
  }
  // 2) transpose + convert weights
  {
    int n = D_MODEL * D_HEAD;
    int grid = (n + 255) / 256;
    cvt_t_kernel<<<grid, 256, 0, stream>>>(w_q, wqT, D_MODEL, D_HEAD);
    cvt_t_kernel<<<grid, 256, 0, stream>>>(w_k, wkT, D_MODEL, D_HEAD);
    cvt_t_kernel<<<grid, 256, 0, stream>>>(w_v, wvT, D_MODEL, D_HEAD);
    cvt_t_kernel<<<grid, 256, 0, stream>>>(w_o, woT, D_HEAD, D_MODEL);
  }
  // 3) projections: (8192/32)*(128/32) = 1024 waves, 8 waves/block -> 128 blocks
  {
    dim3 grid(128), block(256);
    proj_kernel<0><<<grid, block, 0, stream>>>(qbf, wqT, qp);
    proj_kernel<0><<<grid, block, 0, stream>>>(kbf, wkT, kp);
    proj_kernel<1><<<grid, block, 0, stream>>>(vbf, wvT, vpT);
  }
  // 4) flash attention: one block per 16-row query tile
  flash_kernel<<<S_LEN / 16, NWARPS * 32, 0, stream>>>(qp, kp, vpT, obf);

  // 5) output projection: (8192/32)*(1024/32) = 8192 waves -> 1024 blocks
  out_gemm<<<1024, 256, 0, stream>>>(obf, woT, out);
}
